// RolandLayer_27410481283214
// MI455X (gfx1250) — compile-verified, matrix operand order checked by reading, not run on hardware
//
#include <hip/hip_runtime.h>
#include <math.h>

typedef __attribute__((ext_vector_type(16))) _Float16 v16h;
typedef __attribute__((ext_vector_type(8)))  _Float16 v8h;
typedef __attribute__((ext_vector_type(8)))  float    v8f;

#define N_NODES 50000
#define N_EDGES 800000
#define DIM     128
#define EDIM    32
#define NTILES  (N_NODES / 16)   // 3125
#define ETILES  (N_EDGES / 16)   // 50000

// ---- workspace layout (bytes, all offsets multiple of 256) ----
#define OFF_WMSG   0u                                   // 160*128 f16, B-swizzled
#define OFF_WSKIP  (OFF_WMSG  + 160u*128u*2u)
#define OFF_WZ     (OFF_WSKIP + 128u*128u*2u)
#define OFF_WR     (OFF_WZ    + 256u*128u*2u)
#define OFF_WH     (OFF_WR    + 256u*128u*2u)
#define OFF_NF16   (OFF_WH    + 256u*128u*2u)           // N*128 f16 row-major
#define OFF_ST16   (OFF_NF16  + (unsigned)N_NODES*DIM*2u)
#define OFF_H16    (OFF_ST16  + (unsigned)N_NODES*DIM*2u)
#define OFF_RS16   (OFF_H16   + (unsigned)N_NODES*DIM*2u)
#define OFF_AGG    (OFF_RS16  + (unsigned)N_NODES*DIM*2u)  // N*128 f32
#define OFF_ZBUF   (OFF_AGG   + (unsigned)N_NODES*DIM*4u)  // N*128 f32
// total ~102.7 MB

// ---------------- WMMA helpers ----------------

__device__ __forceinline__ v8f wmma16(v16h a, v16h b, v8f c) {
  // D = A(16x32 f16) x B(32x16 f16) + C(16x16 f32)
  return __builtin_amdgcn_wmma_f32_16x16x32_f16(
      /*neg_a=*/false, a, /*neg_b=*/false, b,
      /*c_mod=*/(short)0, c, /*reuse_a=*/false, /*reuse_b=*/false);
}

// A fragment from a row-major f16 row (32-K window starting at `row`).
// Per ISA layout, lane (lo/hi half) needs halfs [8*hi,8*hi+8) and [16+8*hi,24+8*hi)
// -> two contiguous b128 loads.
__device__ __forceinline__ v16h load_a_frag(const _Float16* __restrict__ row, int hi) {
  v8h lo = *(const v8h*)(row + hi * 8);
  v8h hh = *(const v8h*)(row + 16 + hi * 8);
  v16h r;
#pragma unroll
  for (int i = 0; i < 8; ++i) { r[i] = lo[i]; r[8 + i] = hh[i]; }
  return r;
}

// B fragment from a pre-swizzled weight segment: layout [kt][col][32 slots],
// lane_lo reads slots 0..15, lane_hi reads slots 16..31 (one v16h load).
__device__ __forceinline__ v16h load_b_frag(const _Float16* __restrict__ wseg,
                                            int kt, int n, int hi) {
  return *(const v16h*)(wseg + ((((kt << 7) + n) << 5) + (hi ? 16 : 0)));
}

__device__ __forceinline__ float sigmoidf_(float x) {
  return 1.0f / (1.0f + __expf(-x));
}

// ---------------- conversion / swizzle kernels ----------------

// f32 [K][128] -> f16 swizzled for B fragments: dst[((kt*128+c)*32)+slot]
__global__ void k_cvt_weight(const float* __restrict__ src,
                             _Float16* __restrict__ dst, int K) {
  int idx = blockIdx.x * blockDim.x + threadIdx.x;
  if (idx >= K * DIM) return;
  int k = idx >> 7, c = idx & 127;
  int kt = k >> 5, kk = k & 31;
  int grp = kk >> 3;
  int slot = kk + ((grp == 1) ? 8 : (grp == 2) ? -8 : 0);
  dst[(((kt << 7) + c) << 5) + slot] = (_Float16)src[idx];
}

__global__ void k_cvt_pair(const float* __restrict__ a, const float* __restrict__ b,
                           _Float16* __restrict__ a16, _Float16* __restrict__ b16, int n) {
  int i = blockIdx.x * blockDim.x + threadIdx.x;
  if (i < n) { a16[i] = (_Float16)a[i]; b16[i] = (_Float16)b[i]; }
}

__global__ void k_relu_h(const float* __restrict__ agg,
                         _Float16* __restrict__ h16, int n) {
  int i = blockIdx.x * blockDim.x + threadIdx.x;
  if (i < n) h16[i] = (_Float16)fmaxf(agg[i], 0.0f);
}

// ---------------- skip-connection init: agg = nf @ W_skip + b_conv + b_skip ----

__global__ void k_skip(const _Float16* __restrict__ nf16,
                       const _Float16* __restrict__ wsk,
                       const float* __restrict__ b_conv,
                       const float* __restrict__ b_skip,
                       float* __restrict__ agg) {
  int wave = threadIdx.x >> 5;
  int tile = blockIdx.x * (blockDim.x >> 5) + wave;
  if (tile >= NTILES) return;
  int lane = threadIdx.x & 31;
  int col = lane & 15, hi = lane >> 4;
  int node0 = tile << 4;

  v8f acc[8];
#pragma unroll
  for (int i = 0; i < 8; ++i) acc[i] = (v8f){0,0,0,0,0,0,0,0};

#pragma unroll
  for (int kt = 0; kt < 4; ++kt) {
    v16h a = load_a_frag(nf16 + (size_t)(node0 + col) * DIM + kt * 32, hi);
#pragma unroll
    for (int nt = 0; nt < 8; ++nt) {
      v16h b = load_b_frag(wsk, kt, nt * 16 + col, hi);
      acc[nt] = wmma16(a, b, acc[nt]);
    }
  }
#pragma unroll
  for (int nt = 0; nt < 8; ++nt) {
    int n = nt * 16 + col;
    float bias = b_conv[n] + b_skip[n];
#pragma unroll
    for (int r = 0; r < 8; ++r)
      agg[(size_t)(node0 + r + hi * 8) * DIM + n] = acc[nt][r] + bias;
  }
}

// ---------------- edge message GEMM + atomic scatter-add ----------------

__global__ void k_edge(const _Float16* __restrict__ nf16,
                       const float* __restrict__ ef,
                       const int* __restrict__ ei,
                       const _Float16* __restrict__ wmsg,
                       float* __restrict__ agg) {
  int wave = threadIdx.x >> 5;
  int tile = blockIdx.x * (blockDim.x >> 5) + wave;
  if (tile >= ETILES) return;
  int lane = threadIdx.x & 31;
  int col = lane & 15, hi = lane >> 4;
  int e0 = tile << 4;
  int em = e0 + col;            // this lane's A-row edge
  int src = ei[em];             // source node gather index

  v8f acc[8];
#pragma unroll
  for (int i = 0; i < 8; ++i) acc[i] = (v8f){0,0,0,0,0,0,0,0};

  // K-tiles 0..3: gathered node features (f16, L2-hot)
#pragma unroll
  for (int kt = 0; kt < 4; ++kt) {
    v16h a = load_a_frag(nf16 + (size_t)src * DIM + kt * 32, hi);
#pragma unroll
    for (int nt = 0; nt < 8; ++nt) {
      v16h b = load_b_frag(wmsg, kt, nt * 16 + col, hi);
      acc[nt] = wmma16(a, b, acc[nt]);
    }
  }
  // K-tile 4: edge features, f32 -> f16 in-register
  {
    const float4* p = (const float4*)(ef + (size_t)em * EDIM);
    float4 f0 = p[hi * 2 + 0];
    float4 f1 = p[hi * 2 + 1];
    float4 f2 = p[4 + hi * 2 + 0];
    float4 f3 = p[4 + hi * 2 + 1];
    v16h a;
    a[0]=(_Float16)f0.x; a[1]=(_Float16)f0.y; a[2]=(_Float16)f0.z; a[3]=(_Float16)f0.w;
    a[4]=(_Float16)f1.x; a[5]=(_Float16)f1.y; a[6]=(_Float16)f1.z; a[7]=(_Float16)f1.w;
    a[8]=(_Float16)f2.x; a[9]=(_Float16)f2.y; a[10]=(_Float16)f2.z; a[11]=(_Float16)f2.w;
    a[12]=(_Float16)f3.x; a[13]=(_Float16)f3.y; a[14]=(_Float16)f3.z; a[15]=(_Float16)f3.w;
#pragma unroll
    for (int nt = 0; nt < 8; ++nt) {
      v16h b = load_b_frag(wmsg, 4, nt * 16 + col, hi);
      acc[nt] = wmma16(a, b, acc[nt]);
    }
  }
  // scatter-add rows to agg[dst] (f32 atomics, device scope, L2-resident)
#pragma unroll
  for (int r = 0; r < 8; ++r) {
    int dst = ei[N_EDGES + e0 + r + hi * 8];
    float* rowp = agg + (size_t)dst * DIM;
#pragma unroll
    for (int nt = 0; nt < 8; ++nt) {
      __hip_atomic_fetch_add(rowp + nt * 16 + col, acc[nt][r],
                             __ATOMIC_RELAXED, __HIP_MEMORY_SCOPE_AGENT);
    }
  }
}

// ---------------- GRU gates: R (store R*state f16) and Z (store f32) ----------

__global__ void k_zr(const _Float16* __restrict__ h16,
                     const _Float16* __restrict__ st16,
                     const float* __restrict__ states,
                     const _Float16* __restrict__ wz,
                     const _Float16* __restrict__ wr,
                     const float* __restrict__ bz,
                     const float* __restrict__ br,
                     float* __restrict__ Zbuf,
                     _Float16* __restrict__ rs16) {
  int wave = threadIdx.x >> 5;
  int tile = blockIdx.x * (blockDim.x >> 5) + wave;
  if (tile >= NTILES) return;
  int lane = threadIdx.x & 31;
  int col = lane & 15, hi = lane >> 4;
  int node0 = tile << 4;

  v8f acc[8];
  // ---- R gate ----
#pragma unroll
  for (int i = 0; i < 8; ++i) acc[i] = (v8f){0,0,0,0,0,0,0,0};
#pragma unroll
  for (int kt = 0; kt < 8; ++kt) {
    const _Float16* arow = (kt < 4)
        ? h16  + (size_t)(node0 + col) * DIM + kt * 32
        : st16 + (size_t)(node0 + col) * DIM + (kt - 4) * 32;
    v16h a = load_a_frag(arow, hi);
#pragma unroll
    for (int nt = 0; nt < 8; ++nt) {
      v16h b = load_b_frag(wr, kt, nt * 16 + col, hi);
      acc[nt] = wmma16(a, b, acc[nt]);
    }
  }
#pragma unroll
  for (int nt = 0; nt < 8; ++nt) {
    int n = nt * 16 + col;
    float bb = br[n];
#pragma unroll
    for (int r = 0; r < 8; ++r) {
      size_t row = (size_t)(node0 + r + hi * 8);
      float rv = sigmoidf_(acc[nt][r] + bb);
      rs16[row * DIM + n] = (_Float16)(rv * states[row * DIM + n]);
    }
  }
  // ---- Z gate ----
#pragma unroll
  for (int i = 0; i < 8; ++i) acc[i] = (v8f){0,0,0,0,0,0,0,0};
#pragma unroll
  for (int kt = 0; kt < 8; ++kt) {
    const _Float16* arow = (kt < 4)
        ? h16  + (size_t)(node0 + col) * DIM + kt * 32
        : st16 + (size_t)(node0 + col) * DIM + (kt - 4) * 32;
    v16h a = load_a_frag(arow, hi);
#pragma unroll
    for (int nt = 0; nt < 8; ++nt) {
      v16h b = load_b_frag(wz, kt, nt * 16 + col, hi);
      acc[nt] = wmma16(a, b, acc[nt]);
    }
  }
#pragma unroll
  for (int nt = 0; nt < 8; ++nt) {
    int n = nt * 16 + col;
    float bb = bz[n];
#pragma unroll
    for (int r = 0; r < 8; ++r) {
      size_t row = (size_t)(node0 + r + hi * 8);
      Zbuf[row * DIM + n] = sigmoidf_(acc[nt][r] + bb);
    }
  }
}

// ---------------- H_tilde GEMM + gated combine -> out ----------------

__global__ void k_final(const _Float16* __restrict__ h16,
                        const _Float16* __restrict__ rs16,
                        const float* __restrict__ states,
                        const _Float16* __restrict__ wh,
                        const float* __restrict__ bh,
                        const float* __restrict__ Zbuf,
                        float* __restrict__ out) {
  int wave = threadIdx.x >> 5;
  int tile = blockIdx.x * (blockDim.x >> 5) + wave;
  if (tile >= NTILES) return;
  int lane = threadIdx.x & 31;
  int col = lane & 15, hi = lane >> 4;
  int node0 = tile << 4;

  v8f acc[8];
#pragma unroll
  for (int i = 0; i < 8; ++i) acc[i] = (v8f){0,0,0,0,0,0,0,0};
#pragma unroll
  for (int kt = 0; kt < 8; ++kt) {
    const _Float16* arow = (kt < 4)
        ? h16  + (size_t)(node0 + col) * DIM + kt * 32
        : rs16 + (size_t)(node0 + col) * DIM + (kt - 4) * 32;
    v16h a = load_a_frag(arow, hi);
#pragma unroll
    for (int nt = 0; nt < 8; ++nt) {
      v16h b = load_b_frag(wh, kt, nt * 16 + col, hi);
      acc[nt] = wmma16(a, b, acc[nt]);
    }
  }
#pragma unroll
  for (int nt = 0; nt < 8; ++nt) {
    int n = nt * 16 + col;
    float bb = bh[n];
#pragma unroll
    for (int r = 0; r < 8; ++r) {
      size_t row = (size_t)(node0 + r + hi * 8);
      float ht = tanhf(acc[nt][r] + bb);
      float z  = Zbuf[row * DIM + n];
      float s  = states[row * DIM + n];
      out[row * DIM + n] = z * s + (1.0f - z) * ht;
    }
  }
}

// ---------------- host launcher ----------------

extern "C" void kernel_launch(void* const* d_in, const int* in_sizes, int n_in,
                              void* d_out, int out_size, void* d_ws, size_t ws_size,
                              hipStream_t stream) {
  const float* node_feature = (const float*)d_in[0];   // [N,128]
  const float* edge_feature = (const float*)d_in[1];   // [E,32]
  const float* node_states  = (const float*)d_in[2];   // [N,128]
  const int*   edge_index   = (const int*)  d_in[3];   // [2,E]
  const float* W_msg  = (const float*)d_in[4];         // [160,128]
  const float* b_conv = (const float*)d_in[5];         // [128]
  const float* W_skip = (const float*)d_in[6];         // [128,128]
  const float* b_skip = (const float*)d_in[7];         // [128]
  const float* Wz = (const float*)d_in[8];             // [256,128]
  const float* bz = (const float*)d_in[9];
  const float* Wr = (const float*)d_in[10];
  const float* br = (const float*)d_in[11];
  const float* Wh = (const float*)d_in[12];
  const float* bh = (const float*)d_in[13];
  float* out = (float*)d_out;                          // [N,128]

  char* ws = (char*)d_ws;
  _Float16* wmsg16  = (_Float16*)(ws + OFF_WMSG);
  _Float16* wskip16 = (_Float16*)(ws + OFF_WSKIP);
  _Float16* wz16    = (_Float16*)(ws + OFF_WZ);
  _Float16* wr16    = (_Float16*)(ws + OFF_WR);
  _Float16* wh16    = (_Float16*)(ws + OFF_WH);
  _Float16* nf16    = (_Float16*)(ws + OFF_NF16);
  _Float16* st16    = (_Float16*)(ws + OFF_ST16);
  _Float16* h16     = (_Float16*)(ws + OFF_H16);
  _Float16* rs16    = (_Float16*)(ws + OFF_RS16);
  float*    agg     = (float*)   (ws + OFF_AGG);
  float*    Zbuf    = (float*)   (ws + OFF_ZBUF);

  // 1) weight conversion + B-fragment swizzle
  k_cvt_weight<<<(160 * DIM + 255) / 256, 256, 0, stream>>>(W_msg,  wmsg16,  160);
  k_cvt_weight<<<(128 * DIM + 255) / 256, 256, 0, stream>>>(W_skip, wskip16, 128);
  k_cvt_weight<<<(256 * DIM + 255) / 256, 256, 0, stream>>>(Wz, wz16, 256);
  k_cvt_weight<<<(256 * DIM + 255) / 256, 256, 0, stream>>>(Wr, wr16, 256);
  k_cvt_weight<<<(256 * DIM + 255) / 256, 256, 0, stream>>>(Wh, wh16, 256);

  // 2) node features / states -> f16 row-major
  int nelem = N_NODES * DIM;
  k_cvt_pair<<<(nelem + 255) / 256, 256, 0, stream>>>(node_feature, node_states,
                                                      nf16, st16, nelem);

  // 3) agg = nf @ W_skip + b_conv + b_skip   (initializes scatter target)
  k_skip<<<(NTILES + 7) / 8, 256, 0, stream>>>(nf16, wskip16, b_conv, b_skip, agg);

  // 4) agg += scatter_add(concat(nf[src], ef) @ W_msg)
  k_edge<<<ETILES / 8, 256, 0, stream>>>(nf16, edge_feature, edge_index, wmsg16, agg);

  // 5) h = relu(agg) -> f16
  k_relu_h<<<(nelem + 255) / 256, 256, 0, stream>>>(agg, h16, nelem);

  // 6) R, Z gates
  k_zr<<<(NTILES + 7) / 8, 256, 0, stream>>>(h16, st16, node_states,
                                             wz16, wr16, bz, br, Zbuf, rs16);

  // 7) H_tilde + gated combine
  k_final<<<(NTILES + 7) / 8, 256, 0, stream>>>(h16, rs16, node_states,
                                                wh16, bh, Zbuf, out);
}
